// FC_quant_42769284333814
// MI455X (gfx1250) — matrix operand order repaired and checked by reading, not compiled, not run on hardware
//
#include <hip/hip_runtime.h>

// ---------------------------------------------------------------------------
// Z = X @ sign(W) + bias   (X:[B,K] f32, W:[K,N] f32, bias:[N] f32, Z:[B,N] f32)
//
// Plan (MI455X):
//   pass 1a: X f32 -> bf16 (RNE) into ws            [memory-bound, one shot]
//   pass 1b: W f32 -> sign bf16 {-1,0,+1}, TRANSPOSED to [N,K] into ws
//   pass 2 : bf16 GEMM via V_WMMA_F32_16X16X32_BF16, tiles staged with
//            global_load_async_to_lds_b128 (ASYNCcnt), double-buffered LDS.
// Both bf16 operands total 64 MB -> fully L2-resident (192 MB), so K-tile
// re-reads never touch HBM. sign(W) in bf16 is exact; accumulate in f32.
// Falls back to a fused single-pass kernel if ws_size < 64 MB.
// ---------------------------------------------------------------------------

typedef __attribute__((ext_vector_type(16))) __bf16 v16bf;
typedef __attribute__((ext_vector_type(8)))  float  v8f;

#define TILE_M 128
#define TILE_N 128
#define TILE_KB 64   // K-depth per LDS tile (big kernel)
#define TILE_KS 32   // K-depth (fallback kernel)

__device__ __forceinline__ unsigned short f32_to_bf16_bits(float f) {
    unsigned u = __builtin_bit_cast(unsigned, f);
    u += 0x7FFFu + ((u >> 16) & 1u);           // round-to-nearest-even
    return (unsigned short)(u >> 16);
}

__device__ __forceinline__ unsigned short sign_bf16_bits(float w) {
    return (w > 0.0f) ? (unsigned short)0x3F80u
         : (w < 0.0f) ? (unsigned short)0xBF80u
                      : (unsigned short)0x0000u;
}

__device__ __forceinline__ void cvt4_bf16(const float4& f, unsigned short* s) {
    s[0] = f32_to_bf16_bits(f.x); s[1] = f32_to_bf16_bits(f.y);
    s[2] = f32_to_bf16_bits(f.z); s[3] = f32_to_bf16_bits(f.w);
}

__device__ __forceinline__ void sign4_bf16(const float4& f, unsigned short* s) {
    s[0] = sign_bf16_bits(f.x); s[1] = sign_bf16_bits(f.y);
    s[2] = sign_bf16_bits(f.z); s[3] = sign_bf16_bits(f.w);
}

union FragU { v16bf v; uint4 q[2]; };

// ---- CDNA5 async global->LDS copy (ASYNCcnt), 16B per lane ----------------
template <int OFF>
__device__ __forceinline__ void async_copy_b128(unsigned lds_addr, const void* gaddr) {
    asm volatile("global_load_async_to_lds_b128 %0, %1, off offset:%2"
                 :: "v"(lds_addr), "v"(gaddr), "n"(OFF) : "memory");
}
__device__ __forceinline__ void wait_asynccnt0() {
    asm volatile("s_wait_asynccnt 0x0" ::: "memory");
}
// Low 32 bits of a generic LDS address are the LDS byte offset (ISA 10.2).
__device__ __forceinline__ unsigned lds_off(const void* p) {
    return (unsigned)(unsigned long long)p;
}

// ===========================================================================
// Pass 1a: elementwise X f32 -> bf16 (8 elems/thread)
// ===========================================================================
__global__ __launch_bounds__(256)
void convert_X_bf16(const float* __restrict__ X, unsigned short* __restrict__ Xbf) {
    const size_t i = ((size_t)blockIdx.x * 256 + threadIdx.x) * 8;
    const float4* p = (const float4*)(X + i);
    float4 f0 = p[0], f1 = p[1];
    union { unsigned short s[8]; uint4 q; } pk;
    cvt4_bf16(f0, pk.s + 0);
    cvt4_bf16(f1, pk.s + 4);
    *(uint4*)(Xbf + i) = pk.q;
}

// ===========================================================================
// Pass 1b: W[K,N] f32 -> sign bf16, transposed to Wt[N,K] via LDS 64x64 tile
// ===========================================================================
__global__ __launch_bounds__(256)
void sign_transpose_W(const float* __restrict__ W, unsigned short* __restrict__ Wt,
                      int K, int N) {
    __shared__ unsigned short tile[64 * 64];   // [n_local][k_local]
    const int t     = threadIdx.x;
    const int kBase = blockIdx.x * 64;
    const int nBase = blockIdx.y * 64;

    {   // coalesced read of W rows, sign, scatter-transpose into LDS
        const int r  = t >> 2;            // local k: 0..63
        const int c4 = (t & 3) * 16;      // local n start
        const float4* wp = (const float4*)(W + (size_t)(kBase + r) * N + nBase + c4);
        float4 g0 = wp[0], g1 = wp[1], g2 = wp[2], g3 = wp[3];
        unsigned short sg[16];
        sign4_bf16(g0, sg + 0); sign4_bf16(g1, sg + 4);
        sign4_bf16(g2, sg + 8); sign4_bf16(g3, sg + 12);
#pragma unroll
        for (int i = 0; i < 16; ++i)
            tile[(c4 + i) * 64 + r] = sg[i];
    }
    __syncthreads();
    {   // coalesced write of Wt rows
        const int n  = t >> 2;            // local n
        const int k4 = (t & 3) * 16;      // local k start
        uint4*       dst = (uint4*)(Wt + (size_t)(nBase + n) * K + kBase + k4);
        const uint4* src = (const uint4*)(tile + n * 64 + k4);
        dst[0] = src[0];
        dst[1] = src[1];
    }
}

// ===========================================================================
// Pass 2: bf16 GEMM, async-LDS staged, double-buffered.
//   Xbf : [B,K] bf16 row-major      Wt : [N,K] bf16 row-major (pre-signed)
// ===========================================================================
__global__ __launch_bounds__(256)
void bnn_gemm_bf16_async(const unsigned short* __restrict__ Xbf,
                         const unsigned short* __restrict__ Wt,
                         const float* __restrict__ bias,
                         float* __restrict__ Z,
                         int K, int N) {
    __shared__ unsigned short ldsA[2][TILE_M * TILE_KB];  // [m][k], 128B rows
    __shared__ unsigned short ldsB[2][TILE_N * TILE_KB];  // [n][k], 128B rows

    const int t     = threadIdx.x;
    const int lane  = t & 31;
    const int wave  = t >> 5;
    const int waveM = wave & 3;     // 4 waves along M
    const int waveN = wave >> 2;    // 2 waves along N
    const int lr    = lane & 15;
    const int h     = lane >> 4;

    const int mBase = blockIdx.x * TILE_M;
    const int nBase = blockIdx.y * TILE_N;

    // staging: 2 threads per 128B tile row; each moves 64B = 4x b128 async
    const int sr = t >> 1;                 // tile row 0..127
    const int sc = (t & 1) * 32;           // bf16 col offset 0 / 32

    auto issue_tiles = [&](int buf, int kt) {
        const unsigned la = lds_off(&ldsA[buf][sr * TILE_KB + sc]);
        const void*    ga = Xbf + (size_t)(mBase + sr) * K + kt + sc;
        async_copy_b128<0>(la, ga);  async_copy_b128<16>(la, ga);
        async_copy_b128<32>(la, ga); async_copy_b128<48>(la, ga);
        const unsigned lb = lds_off(&ldsB[buf][sr * TILE_KB + sc]);
        const void*    gb = Wt + (size_t)(nBase + sr) * K + kt + sc;
        async_copy_b128<0>(lb, gb);  async_copy_b128<16>(lb, gb);
        async_copy_b128<32>(lb, gb); async_copy_b128<48>(lb, gb);
    };

    v8f acc[2][4];
#pragma unroll
    for (int mf = 0; mf < 2; ++mf)
#pragma unroll
        for (int nf = 0; nf < 4; ++nf)
            acc[mf][nf] = (v8f){0.f, 0.f, 0.f, 0.f, 0.f, 0.f, 0.f, 0.f};

    const int nk = K / TILE_KB;
    issue_tiles(0, 0);

    for (int i = 0; i < nk; ++i) {
        const int buf = i & 1;
        wait_asynccnt0();     // own async loads for ldsX[buf] done
        __syncthreads();      // -> everyone's loads done; prev reads of buf^1 done
        if (i + 1 < nk)
            issue_tiles(buf ^ 1, (i + 1) * TILE_KB);

        // two K=32 sub-steps over the 64-deep tile
#pragma unroll
        for (int kk = 0; kk < TILE_KB; kk += 32) {
            FragU a[2];
#pragma unroll
            for (int mf = 0; mf < 2; ++mf) {
                const int m_local = waveM * 32 + mf * 16 + lr;
                const uint4* pa =
                    (const uint4*)(&ldsA[buf][m_local * TILE_KB + kk + 8 * h]);
                a[mf].q[0] = pa[0];   // K = kk+8h .. kk+8h+7
                a[mf].q[1] = pa[2];   // K = kk+16+8h .. kk+23+8h
            }
            FragU b[4];
#pragma unroll
            for (int nf = 0; nf < 4; ++nf) {
                const int n_local = waveN * 64 + nf * 16 + lr;
                const uint4* pb =
                    (const uint4*)(&ldsB[buf][n_local * TILE_KB + kk + 8 * h]);
                b[nf].q[0] = pb[0];   // K = kk+8h .. kk+8h+7
                b[nf].q[1] = pb[2];   // K = kk+16+8h .. kk+23+8h
            }
#pragma unroll
            for (int mf = 0; mf < 2; ++mf)
#pragma unroll
                for (int nf = 0; nf < 4; ++nf)
                    acc[mf][nf] = __builtin_amdgcn_wmma_f32_16x16x32_bf16(
                        false, a[mf].v, false, b[nf].v,
                        (short)0, acc[mf][nf], false, false);
        }
    }

    // epilogue: C/D layout VGPR j -> M = j + 8h, N = lane&15 ; add bias
#pragma unroll
    for (int mf = 0; mf < 2; ++mf) {
#pragma unroll
        for (int nf = 0; nf < 4; ++nf) {
            const int gn = nBase + waveN * 64 + nf * 16 + lr;
            const float bv = bias[gn];
#pragma unroll
            for (int j = 0; j < 8; ++j) {
                const int gm = mBase + waveM * 32 + mf * 16 + j + 8 * h;
                Z[(size_t)gm * N + gn] = acc[mf][nf][j] + bv;
            }
        }
    }
}

// ===========================================================================
// Fallback: fused single-pass kernel (no workspace needed)
// ===========================================================================
__global__ __launch_bounds__(256)
void bnn_gemm_bf16_fused(const float* __restrict__ X,
                         const float* __restrict__ W,
                         const float* __restrict__ bias,
                         float* __restrict__ Z,
                         int K, int N) {
    __shared__ unsigned short ldsA[TILE_M * TILE_KS];
    __shared__ unsigned short ldsB[TILE_N * TILE_KS];

    const int t     = threadIdx.x;
    const int lane  = t & 31;
    const int wave  = t >> 5;
    const int waveM = wave & 3;
    const int waveN = wave >> 2;
    const int lr    = lane & 15;
    const int h     = lane >> 4;
    const int mBase = blockIdx.x * TILE_M;
    const int nBase = blockIdx.y * TILE_N;

    v8f acc[2][4];
#pragma unroll
    for (int mf = 0; mf < 2; ++mf)
#pragma unroll
        for (int nf = 0; nf < 4; ++nf)
            acc[mf][nf] = (v8f){0.f, 0.f, 0.f, 0.f, 0.f, 0.f, 0.f, 0.f};

    const int xr = t >> 1, xc = (t & 1) * 16;
    const int wk = t >> 3, wc = (t & 7) * 16;

    for (int kt = 0; kt < K; kt += TILE_KS) {
        __syncthreads();
        {
            const float4* xp = (const float4*)(X + (size_t)(mBase + xr) * K + kt + xc);
            float4 f0 = xp[0], f1 = xp[1], f2 = xp[2], f3 = xp[3];
            union { unsigned short s[16]; uint4 q[2]; } pk;
            cvt4_bf16(f0, pk.s + 0);  cvt4_bf16(f1, pk.s + 4);
            cvt4_bf16(f2, pk.s + 8);  cvt4_bf16(f3, pk.s + 12);
            uint4* dst = (uint4*)(ldsA + xr * TILE_KS + xc);
            dst[0] = pk.q[0]; dst[1] = pk.q[1];
        }
        {
            const float4* wp = (const float4*)(W + (size_t)(kt + wk) * N + nBase + wc);
            float4 g0 = wp[0], g1 = wp[1], g2 = wp[2], g3 = wp[3];
            unsigned short sg[16];
            sign4_bf16(g0, sg + 0);  sign4_bf16(g1, sg + 4);
            sign4_bf16(g2, sg + 8);  sign4_bf16(g3, sg + 12);
#pragma unroll
            for (int i = 0; i < 16; ++i)
                ldsB[(wc + i) * TILE_KS + wk] = sg[i];
        }
        __syncthreads();

        FragU a[2];
#pragma unroll
        for (int mf = 0; mf < 2; ++mf) {
            const int m_local = waveM * 32 + mf * 16 + lr;
            const uint4* pa = (const uint4*)(ldsA + m_local * TILE_KS + 8 * h);
            a[mf].q[0] = pa[0]; a[mf].q[1] = pa[2];
        }
        FragU b[4];
#pragma unroll
        for (int nf = 0; nf < 4; ++nf) {
            const int n_local = waveN * 64 + nf * 16 + lr;
            const uint4* pb = (const uint4*)(ldsB + n_local * TILE_KS + 8 * h);
            b[nf].q[0] = pb[0]; b[nf].q[1] = pb[2];
        }
#pragma unroll
        for (int mf = 0; mf < 2; ++mf)
#pragma unroll
            for (int nf = 0; nf < 4; ++nf)
                acc[mf][nf] = __builtin_amdgcn_wmma_f32_16x16x32_bf16(
                    false, a[mf].v, false, b[nf].v,
                    (short)0, acc[mf][nf], false, false);
    }

#pragma unroll
    for (int mf = 0; mf < 2; ++mf) {
#pragma unroll
        for (int nf = 0; nf < 4; ++nf) {
            const int gn = nBase + waveN * 64 + nf * 16 + lr;
            const float bv = bias[gn];
#pragma unroll
            for (int j = 0; j < 8; ++j) {
                const int gm = mBase + waveM * 32 + mf * 16 + j + 8 * h;
                Z[(size_t)gm * N + gn] = acc[mf][nf][j] + bv;
            }
        }
    }
}

// ===========================================================================
extern "C" void kernel_launch(void* const* d_in, const int* in_sizes, int n_in,
                              void* d_out, int out_size, void* d_ws, size_t ws_size,
                              hipStream_t stream) {
    (void)n_in; (void)out_size;

    const float* X    = (const float*)d_in[0];
    const float* W    = (const float*)d_in[1];
    const float* bias = (const float*)d_in[2];
    float*       Z    = (float*)d_out;

    const int N  = in_sizes[2];          // D_OUT from bias length
    const int K  = in_sizes[1] / N;      // D_IN
    const int Bm = in_sizes[0] / K;      // batch

    const size_t xbf_bytes = (size_t)Bm * K * sizeof(unsigned short);
    const size_t wt_bytes  = (size_t)K  * N * sizeof(unsigned short);

    if (ws_size >= xbf_bytes + wt_bytes) {
        unsigned short* Xbf = (unsigned short*)d_ws;
        unsigned short* Wt  = (unsigned short*)((char*)d_ws + xbf_bytes);

        const size_t xelems = (size_t)Bm * K;
        convert_X_bf16<<<dim3((unsigned)(xelems / (256 * 8))), 256, 0, stream>>>(X, Xbf);
        sign_transpose_W<<<dim3(K / 64, N / 64), 256, 0, stream>>>(W, Wt, K, N);

        dim3 grid(Bm / TILE_M, N / TILE_N);
        bnn_gemm_bf16_async<<<grid, 256, 0, stream>>>(Xbf, Wt, bias, Z, K, N);
    } else {
        dim3 grid(Bm / TILE_M, N / TILE_N);
        bnn_gemm_bf16_fused<<<grid, 256, 0, stream>>>(X, W, bias, Z, K, N);
    }
}
